// GLU_46815143526760
// MI455X (gfx1250) — matrix operand order repaired and checked by reading, not compiled, not run on hardware
//
#include <hip/hip_runtime.h>
#include <hip/hip_bf16.h>
#include <math.h>

// Problem dims (match reference)
#define E_ 16
#define T_ 2048
#define H_ 1024
#define F_ 4096

typedef __attribute__((ext_vector_type(16))) __bf16 bf16x16;
typedef __attribute__((ext_vector_type(8)))  __bf16 bf16x8;
typedef __attribute__((ext_vector_type(8)))  float  f32x8;
typedef __attribute__((ext_vector_type(4)))  float  f32x4;

// LDS tile geometry: K-slab of 64 bf16 per row = 128B data, padded to 144B
// (36 dword banks -> 36*r mod 64 distinct for r=0..15: conflict-free frags)
#define ROWB 144

// GEMM1 stage: A 256x64, B1 64x64, B2 64x64
#define G1_A_OFS   0
#define G1_B1_OFS  (256 * ROWB)              // 36864
#define G1_B2_OFS  (G1_B1_OFS + 64 * ROWB)   // 46080
#define G1_STAGE   (G1_B2_OFS + 64 * ROWB)   // 55296
// GEMM2 stage: A 256x64, B 64x64
#define G2_A_OFS   0
#define G2_B_OFS   (256 * ROWB)              // 36864
#define G2_STAGE   (G2_B_OFS + 64 * ROWB)    // 46080

// ---------------- CDNA5 async copy helpers ----------------
// GVS form: lds_addr (VGPR, low 32 bits of generic LDS address),
// 32-bit byte offset (VGPR), 64-bit base (SGPR pair). Tracked by ASYNCcnt.
__device__ __forceinline__ void async_b128(const char* lds_ptr, const char* gbase, int goff) {
    unsigned lds = (unsigned)(uintptr_t)lds_ptr;
    asm volatile("global_load_async_to_lds_b128 %0, %1, %2"
                 :: "v"(lds), "v"(goff), "s"(gbase) : "memory");
}
__device__ __forceinline__ void wait_async_le12() { asm volatile("s_wait_asynccnt 0xc" ::: "memory"); }
__device__ __forceinline__ void wait_async_le10() { asm volatile("s_wait_asynccnt 0xa" ::: "memory"); }
__device__ __forceinline__ void wait_async_le0()  { asm volatile("s_wait_asynccnt 0x0" ::: "memory"); }

// ---------------- prep kernels ----------------
__global__ __launch_bounds__(256) void cvt_bf16_kernel(const float* __restrict__ src,
                                                       __bf16* __restrict__ dst,
                                                       size_t n) {
    size_t i = ((size_t)blockIdx.x * blockDim.x + threadIdx.x) * 8;
    if (i >= n) return;
    f32x4 a = *(const f32x4*)(src + i);
    f32x4 b = *(const f32x4*)(src + i + 4);
    bf16x8 o;
    o[0] = (__bf16)a[0]; o[1] = (__bf16)a[1]; o[2] = (__bf16)a[2]; o[3] = (__bf16)a[3];
    o[4] = (__bf16)b[0]; o[5] = (__bf16)b[1]; o[6] = (__bf16)b[2]; o[7] = (__bf16)b[3];
    *(bf16x8*)(dst + i) = o;
}

// w2 [E][F][H] fp32 -> w2t [E][H][F] bf16 (32x32 LDS tiles)
__global__ __launch_bounds__(256) void w2_transpose_kernel(const float* __restrict__ w2,
                                                           __bf16* __restrict__ w2t) {
    __shared__ float tile[32][33];
    const int e  = blockIdx.z;
    const int f0 = blockIdx.x * 32;
    const int h0 = blockIdx.y * 32;
    const int tx = threadIdx.x;   // 0..31
    const int ty = threadIdx.y;   // 0..7
    const float* src = w2 + (size_t)e * F_ * H_;
#pragma unroll
    for (int i = 0; i < 4; ++i)
        tile[ty + i * 8][tx] = src[(size_t)(f0 + ty + i * 8) * H_ + (h0 + tx)];
    __syncthreads();
    __bf16* dst = w2t + (size_t)e * H_ * F_;
#pragma unroll
    for (int i = 0; i < 4; ++i)
        dst[(size_t)(h0 + ty + i * 8) * F_ + (f0 + tx)] = (__bf16)tile[tx][ty + i * 8];
}

// ---------------- LDS fragment loaders (ISA VGPR layouts) ----------------
// A (16x32): lane m (hi=0) K 0..7 & 16..23 ; lane m+16 K 8..15 & 24..31
__device__ __forceinline__ bf16x16 lds_frag_a(const char* rowbase, int kk, int hi) {
    bf16x16 v;
    const char* p = rowbase + (kk + hi * 8) * 2;
    ((bf16x8*)&v)[0] = *(const bf16x8*)p;
    ((bf16x8*)&v)[1] = *(const bf16x8*)(p + 32);
    return v;
}
// B (32x16): lane n (hi=0) col n K 0..7 & 16..23 ; lane n+16 K 8..15 & 24..31 (K-contiguous rows)
__device__ __forceinline__ bf16x16 lds_frag_b(const char* rowbase, int kk, int hi) {
    bf16x16 v;
    const char* p = rowbase + (kk + hi * 8) * 2;
    ((bf16x8*)&v)[0] = *(const bf16x8*)p;
    ((bf16x8*)&v)[1] = *(const bf16x8*)(p + 32);
    return v;
}

__device__ __forceinline__ float gelu_tanh(float x) {
    const float c0 = 0.7978845608028654f;  // sqrt(2/pi)
    const float c1 = 0.044715f;
    float x3 = x * x * x;
    return 0.5f * x * (1.0f + tanhf(c0 * (x + c1 * x3)));
}

// ---------------- GEMM1 + GLU ----------------
// block = 256 threads (8 waves). Wave: M=32 (2 tiles), N=64 (4 tiles), both mats.
// WG tile 256(M) x 64(N). grid = (F/64, T/256, E). K staged 64 at a time, double-buffered.
__global__ __launch_bounds__(256) void glu_gemm1_kernel(const __bf16* __restrict__ xb,
                                                        const __bf16* __restrict__ w1b,
                                                        const __bf16* __restrict__ v1b,
                                                        __bf16* __restrict__ hb) {
    extern __shared__ char smem[];
    const int e    = blockIdx.z;
    const int t    = threadIdx.x;
    const int wave = t >> 5;
    const int lane = t & 31;
    const int r    = lane & 15;
    const int hi   = lane >> 4;
    const int m0   = blockIdx.y * 256;
    const int n0   = blockIdx.x * 64;

    const char* abase  = (const char*)(xb  + ((size_t)e * T_ + m0) * H_);
    const char* b1base = (const char*)(w1b + ((size_t)e * F_ + n0) * H_);
    const char* b2base = (const char*)(v1b + ((size_t)e * F_ + n0) * H_);

    // stage issuer: 12 async b128 per thread (A:8, B1:2, B2:2)
    auto issue = [&](int buf, int k0) {
        char* st = smem + buf * G1_STAGE;
        int kb = k0 * 2;
#pragma unroll
        for (int i = 0; i < 8; ++i) {            // A: 2048 chunks (256 rows x 8 segs)
            int c = t + i * 256, row = c >> 3, seg = c & 7;
            async_b128(st + G1_A_OFS + row * ROWB + seg * 16, abase,
                       row * (H_ * 2) + kb + seg * 16);
        }
#pragma unroll
        for (int i = 0; i < 2; ++i) {            // B1/B2: 512 chunks each
            int c = t + i * 256, row = c >> 3, seg = c & 7;
            int go = row * (H_ * 2) + kb + seg * 16;
            async_b128(st + G1_B1_OFS + row * ROWB + seg * 16, b1base, go);
            async_b128(st + G1_B2_OFS + row * ROWB + seg * 16, b2base, go);
        }
    };

    f32x8 acc1[2][4], acc2[2][4];
#pragma unroll
    for (int mt = 0; mt < 2; ++mt)
#pragma unroll
        for (int nt = 0; nt < 4; ++nt) { acc1[mt][nt] = f32x8{0.f}; acc2[mt][nt] = f32x8{0.f}; }

    const int S = H_ / 64;                       // 16 stages
    issue(0, 0);
    for (int s = 0; s < S; ++s) {
        const int cur = s & 1;
        if (s + 1 < S) { issue(cur ^ 1, (s + 1) * 64); wait_async_le12(); }
        else          { wait_async_le0(); }
        __syncthreads();

        const char* st = smem + cur * G1_STAGE;
#pragma unroll
        for (int kk = 0; kk < 64; kk += 32) {
            bf16x16 a[2];
#pragma unroll
            for (int mt = 0; mt < 2; ++mt)
                a[mt] = lds_frag_a(st + G1_A_OFS + (wave * 32 + mt * 16 + r) * ROWB, kk, hi);
#pragma unroll
            for (int nt = 0; nt < 4; ++nt) {
                bf16x16 b1 = lds_frag_b(st + G1_B1_OFS + (nt * 16 + r) * ROWB, kk, hi);
                bf16x16 b2 = lds_frag_b(st + G1_B2_OFS + (nt * 16 + r) * ROWB, kk, hi);
#pragma unroll
                for (int mt = 0; mt < 2; ++mt) {
                    acc1[mt][nt] = __builtin_amdgcn_wmma_f32_16x16x32_bf16(
                        false, a[mt], false, b1, (short)0, acc1[mt][nt], false, false);
                    acc2[mt][nt] = __builtin_amdgcn_wmma_f32_16x16x32_bf16(
                        false, a[mt], false, b2, (short)0, acc2[mt][nt], false, false);
                }
            }
        }
        __syncthreads();
    }

    __bf16* hbase = hb + (size_t)e * T_ * F_;
#pragma unroll
    for (int mt = 0; mt < 2; ++mt)
#pragma unroll
        for (int nt = 0; nt < 4; ++nt) {
            int col = n0 + nt * 16 + r;
#pragma unroll
            for (int j = 0; j < 8; ++j) {
                int row = m0 + wave * 32 + mt * 16 + hi * 8 + j;
                float g = gelu_tanh(acc1[mt][nt][j]) * acc2[mt][nt][j];
                hbase[(size_t)row * F_ + col] = (__bf16)g;
            }
        }
}

// ---------------- GEMM2: out = h @ w2 (w2t is [H][F], K-contiguous) ----------------
// grid = (H/64, T/256, E)
__global__ __launch_bounds__(256) void glu_gemm2_kernel(const __bf16* __restrict__ hb,
                                                        const __bf16* __restrict__ w2t,
                                                        float* __restrict__ out) {
    extern __shared__ char smem[];
    const int e    = blockIdx.z;
    const int t    = threadIdx.x;
    const int wave = t >> 5;
    const int lane = t & 31;
    const int r    = lane & 15;
    const int hi   = lane >> 4;
    const int m0   = blockIdx.y * 256;
    const int n0   = blockIdx.x * 64;

    const char* abase = (const char*)(hb  + ((size_t)e * T_ + m0) * F_);
    const char* bbase = (const char*)(w2t + ((size_t)e * H_ + n0) * F_);

    auto issue = [&](int buf, int k0) {          // 10 async b128 per thread (A:8, B:2)
        char* st = smem + buf * G2_STAGE;
        int kb = k0 * 2;
#pragma unroll
        for (int i = 0; i < 8; ++i) {
            int c = t + i * 256, row = c >> 3, seg = c & 7;
            async_b128(st + G2_A_OFS + row * ROWB + seg * 16, abase,
                       row * (F_ * 2) + kb + seg * 16);
        }
#pragma unroll
        for (int i = 0; i < 2; ++i) {
            int c = t + i * 256, row = c >> 3, seg = c & 7;
            async_b128(st + G2_B_OFS + row * ROWB + seg * 16, bbase,
                       row * (F_ * 2) + kb + seg * 16);
        }
    };

    f32x8 acc[2][4];
#pragma unroll
    for (int mt = 0; mt < 2; ++mt)
#pragma unroll
        for (int nt = 0; nt < 4; ++nt) acc[mt][nt] = f32x8{0.f};

    const int S = F_ / 64;                       // 64 stages
    issue(0, 0);
    for (int s = 0; s < S; ++s) {
        const int cur = s & 1;
        if (s + 1 < S) { issue(cur ^ 1, (s + 1) * 64); wait_async_le10(); }
        else          { wait_async_le0(); }
        __syncthreads();

        const char* st = smem + cur * G2_STAGE;
#pragma unroll
        for (int kk = 0; kk < 64; kk += 32) {
            bf16x16 a[2];
#pragma unroll
            for (int mt = 0; mt < 2; ++mt)
                a[mt] = lds_frag_a(st + G2_A_OFS + (wave * 32 + mt * 16 + r) * ROWB, kk, hi);
#pragma unroll
            for (int nt = 0; nt < 4; ++nt) {
                bf16x16 b = lds_frag_b(st + G2_B_OFS + (nt * 16 + r) * ROWB, kk, hi);
#pragma unroll
                for (int mt = 0; mt < 2; ++mt)
                    acc[mt][nt] = __builtin_amdgcn_wmma_f32_16x16x32_bf16(
                        false, a[mt], false, b, (short)0, acc[mt][nt], false, false);
            }
        }
        __syncthreads();
    }

    float* obase = out + (size_t)e * T_ * H_;
#pragma unroll
    for (int mt = 0; mt < 2; ++mt)
#pragma unroll
        for (int nt = 0; nt < 4; ++nt) {
            int col = n0 + nt * 16 + r;
#pragma unroll
            for (int j = 0; j < 8; ++j) {
                int row = m0 + wave * 32 + mt * 16 + hi * 8 + j;
                obase[(size_t)row * H_ + col] = acc[mt][nt][j];
            }
        }
}

// ---------------- launch ----------------
extern "C" void kernel_launch(void* const* d_in, const int* in_sizes, int n_in,
                              void* d_out, int out_size, void* d_ws, size_t ws_size,
                              hipStream_t stream) {
    (void)in_sizes; (void)n_in; (void)out_size; (void)ws_size;
    const float* x  = (const float*)d_in[0];
    const float* w1 = (const float*)d_in[1];
    const float* v1 = (const float*)d_in[2];
    const float* w2 = (const float*)d_in[3];
    float* out = (float*)d_out;

    const size_t nx = (size_t)E_ * T_ * H_;
    const size_t nw = (size_t)E_ * F_ * H_;

    char* ws = (char*)d_ws;
    __bf16* xb  = (__bf16*)(ws);
    __bf16* w1b = (__bf16*)(ws + nx * 2);
    __bf16* v1b = (__bf16*)(ws + nx * 2 + nw * 2);
    __bf16* w2t = (__bf16*)(ws + nx * 2 + 2 * nw * 2);
    __bf16* hb  = (__bf16*)(ws + nx * 2 + 3 * nw * 2);
    // workspace use: 2*(nx + 3*nw + nh) = 704 MiB

    cvt_bf16_kernel<<<dim3((unsigned)(nx / 8 / 256)), dim3(256), 0, stream>>>(x,  xb,  nx);
    cvt_bf16_kernel<<<dim3((unsigned)(nw / 8 / 256)), dim3(256), 0, stream>>>(w1, w1b, nw);
    cvt_bf16_kernel<<<dim3((unsigned)(nw / 8 / 256)), dim3(256), 0, stream>>>(v1, v1b, nw);
    w2_transpose_kernel<<<dim3(F_ / 32, H_ / 32, E_), dim3(32, 8), 0, stream>>>(w2, w2t);

    glu_gemm1_kernel<<<dim3(F_ / 64, T_ / 256, E_), dim3(256), 2 * G1_STAGE, stream>>>(
        xb, w1b, v1b, hb);
    glu_gemm2_kernel<<<dim3(H_ / 64, T_ / 256, E_), dim3(256), 2 * G2_STAGE, stream>>>(
        hb, w2t, out);
}
